// GCN_35656818492215
// MI455X (gfx1250) — matrix operand order repaired and checked by reading, not compiled, not run on hardware
//
#include <hip/hip_runtime.h>
#include <stddef.h>

typedef __attribute__((ext_vector_type(16))) _Float16 v16h;
typedef __attribute__((ext_vector_type(8)))  float    v8f;

#define N_NODES 100000
#define N_EDGES 3200000
#define N_GRAPHS 256

// ---------------------------------------------------------------------------
// Weight pack: f32 W[K,N] -> f16 packed in WMMA B-operand lane order.
// Layout: [ntile][kchunk][lane(32)][16 halves], 32B per lane -> one v16h load.
// Lane l covers column n = ntile*16 + (l&15), K block kb = (l&16)?16:0,
// halves j=0..15 hold W[kc*32+kb+j][n] (zero if n >= N).
// ---------------------------------------------------------------------------
__global__ void gcn_pack_w(const float* __restrict__ W, _Float16* __restrict__ out,
                           int K, int N, int Nt, int Kc) {
    int tid = blockIdx.x * blockDim.x + threadIdx.x;
    if (tid >= Nt * Kc * 32) return;
    int lane  = tid & 31;
    int rest  = tid >> 5;
    int kc    = rest % Kc;
    int ntile = rest / Kc;
    int n  = (ntile << 4) + (lane & 15);
    int kb = (lane & 16) ? 16 : 0;
    v16h v;
#pragma unroll
    for (int j = 0; j < 16; ++j) {
        int k = (kc << 5) + kb + j;
        v[j] = (n < N) ? (_Float16)W[(size_t)k * N + n] : (_Float16)0.f;
    }
    *(v16h*)(out + (size_t)tid * 16) = v;
}

// ---------------------------------------------------------------------------
// WMMA GEMM: C[M,N] = A[M,K] @ B[K,N] (+bias) (+relu)
// A row-major f32 (K % 32 == 0, M % 16 == 0), B pre-packed f16 (see above).
// One wave computes one 16x16 tile via v_wmma_f32_16x16x32_f16.
// ---------------------------------------------------------------------------
__global__ void gcn_wmma_gemm(const float* __restrict__ A,
                              const _Float16* __restrict__ Bp,
                              const float* __restrict__ bias,
                              float* __restrict__ C,
                              int M, int K, int N, int relu) {
    int wave = blockIdx.x * (blockDim.x >> 5) + (threadIdx.x >> 5);
    int lane = threadIdx.x & 31;
    int Mt = M >> 4;
    int Nt = (N + 15) >> 4;
    int Kc = K >> 5;
    if (wave >= Mt * Nt) return;          // wave-uniform exit

    int mtile = wave % Mt;
    int ntile = wave / Mt;
    int m0 = mtile << 4, n0 = ntile << 4;

    int mrow = lane & 15;
    int hi   = lane >> 4;                 // half-wave: 0/1
    int kbA  = hi << 4;                   // A K-base: 0 / 16

    const float*    Ap = A + (size_t)(m0 + mrow) * K + kbA;
    const _Float16* Bl = Bp + ((size_t)ntile * Kc * 32 + lane) * 16;

    v8f acc = {};
    for (int kc = 0; kc < Kc; ++kc) {
        float4 x0 = *(const float4*)(Ap + 0);
        float4 x1 = *(const float4*)(Ap + 4);
        float4 x2 = *(const float4*)(Ap + 8);
        float4 x3 = *(const float4*)(Ap + 12);
        v16h a;
        a[0]  = (_Float16)x0.x; a[1]  = (_Float16)x0.y;
        a[2]  = (_Float16)x0.z; a[3]  = (_Float16)x0.w;
        a[4]  = (_Float16)x1.x; a[5]  = (_Float16)x1.y;
        a[6]  = (_Float16)x1.z; a[7]  = (_Float16)x1.w;
        a[8]  = (_Float16)x2.x; a[9]  = (_Float16)x2.y;
        a[10] = (_Float16)x2.z; a[11] = (_Float16)x2.w;
        a[12] = (_Float16)x3.x; a[13] = (_Float16)x3.y;
        a[14] = (_Float16)x3.z; a[15] = (_Float16)x3.w;
        v16h b = *(const v16h*)Bl;
        // (neg_a, A, neg_b, B, c_mod, C, reuse_a, reuse_b)
        acc = __builtin_amdgcn_wmma_f32_16x16x32_f16(false, a, false, b,
                                                     (short)0, acc, false, false);
        Ap += 32;
        Bl += 32 * 16;
    }

    int bn = n0 + (lane & 15);
    if (bn < N) {
        float bv = bias ? bias[bn] : 0.f;
#pragma unroll
        for (int j = 0; j < 8; ++j) {
            int row = m0 + j + (hi << 3);
            float v = acc[j] + bv;
            if (relu) v = fmaxf(v, 0.f);
            C[(size_t)row * N + bn] = v;
        }
    }
}

// ---------------------------------------------------------------------------
// Layer 1 projection is rank-1: h[n,f] = x[n] * W1[0,f]  (Fin = 1, Fout = 32)
// ---------------------------------------------------------------------------
__global__ void gcn_l1_proj(const float* __restrict__ x, const float* __restrict__ W1,
                            float* __restrict__ h, int Nn) {
    int i = blockIdx.x * blockDim.x + threadIdx.x;
    if (i >= Nn * 32) return;
    h[i] = x[i >> 5] * W1[i & 31];
}

// ---------------------------------------------------------------------------
// Utility / graph kernels
// ---------------------------------------------------------------------------
__global__ void gcn_fill(float* __restrict__ p, float v, size_t n) {
    size_t i = (size_t)blockIdx.x * blockDim.x + threadIdx.x;
    if (i < n) p[i] = v;
}

__global__ void gcn_deg(const int* __restrict__ dst, float* __restrict__ deg, int E) {
    int e = blockIdx.x * blockDim.x + threadIdx.x;
    if (e < E) unsafeAtomicAdd(deg + dst[e], 1.0f);
}

__global__ void gcn_rsqrt(float* __restrict__ d, int n) {
    int i = blockIdx.x * blockDim.x + threadIdx.x;
    if (i < n) d[i] = rsqrtf(d[i]);
}

// One thread handles 4 features of one edge: agg[dst] += h[src] * dinv[s]*dinv[d]
__global__ void gcn_scatter(const float* __restrict__ h,
                            const int* __restrict__ src,
                            const int* __restrict__ dst,
                            const float* __restrict__ dinv,
                            float* __restrict__ agg,
                            int E, int f4shift, int F) {
    long long tid = (long long)blockIdx.x * blockDim.x + threadIdx.x;
    long long total = (long long)E << f4shift;
    if (tid >= total) return;
    int e  = (int)(tid >> f4shift);
    int fg = ((int)tid & ((1 << f4shift) - 1)) << 2;
    int s = src[e], d = dst[e];
    float norm = dinv[s] * dinv[d];
    const float4 hv = *(const float4*)(h + (size_t)s * F + fg);
    float* base = agg + (size_t)d * F + fg;
    unsafeAtomicAdd(base + 0, hv.x * norm);
    unsafeAtomicAdd(base + 1, hv.y * norm);
    unsafeAtomicAdd(base + 2, hv.z * norm);
    unsafeAtomicAdd(base + 3, hv.w * norm);
}

// In-place: agg = relu(agg + h*dinv^2 + bias)
__global__ void gcn_combine(float* __restrict__ agg, const float* __restrict__ h,
                            const float* __restrict__ dinv,
                            const float* __restrict__ bias,
                            int Nn, int fshift) {
    size_t i = (size_t)blockIdx.x * blockDim.x + threadIdx.x;
    size_t total = (size_t)Nn << fshift;
    if (i >= total) return;
    int n = (int)(i >> fshift);
    int f = (int)i & ((1 << fshift) - 1);
    float di = dinv[n];
    float v = agg[i] + h[i] * di * di + bias[f];
    agg[i] = fmaxf(v, 0.f);
}

// sums[batch[n]][f..f+3] += h[n][f..f+3]   (F fixed at 256)
__global__ void gcn_pool_accum(const float* __restrict__ h,
                               const int* __restrict__ batch,
                               float* __restrict__ sums, int Nn) {
    long long tid = (long long)blockIdx.x * blockDim.x + threadIdx.x;
    if (tid >= (long long)Nn * 64) return;
    int n  = (int)(tid >> 6);
    int fg = ((int)tid & 63) << 2;
    int g = batch[n];
    const float4 hv = *(const float4*)(h + (size_t)n * 256 + fg);
    float* base = sums + (size_t)g * 256 + fg;
    unsafeAtomicAdd(base + 0, hv.x);
    unsafeAtomicAdd(base + 1, hv.y);
    unsafeAtomicAdd(base + 2, hv.z);
    unsafeAtomicAdd(base + 3, hv.w);
}

__global__ void gcn_pool_cnt(const int* __restrict__ batch, float* __restrict__ cnt, int Nn) {
    int n = blockIdx.x * blockDim.x + threadIdx.x;
    if (n < Nn) unsafeAtomicAdd(cnt + batch[n], 1.0f);
}

__global__ void gcn_pool_mean(const float* __restrict__ sums,
                              const float* __restrict__ cnt,
                              float* __restrict__ g) {
    int i = blockIdx.x * blockDim.x + threadIdx.x;
    if (i >= N_GRAPHS * 256) return;
    g[i] = sums[i] / fmaxf(cnt[i >> 8], 1.0f);
}

// ---------------------------------------------------------------------------
// Host-side orchestration
// ---------------------------------------------------------------------------
static inline unsigned cdiv(long long a, int b) { return (unsigned)((a + b - 1) / b); }

static void launch_gemm(const float* A, const _Float16* Bp, const float* bias, float* C,
                        int M, int K, int N, int relu, hipStream_t s) {
    int waves = (M >> 4) * ((N + 15) >> 4);
    gcn_wmma_gemm<<<cdiv(waves, 8), 256, 0, s>>>(A, Bp, bias, C, M, K, N, relu);
}

static void launch_pack(const float* W, _Float16* out, int K, int N, hipStream_t s) {
    int Nt = (N + 15) >> 4, Kc = K >> 5;
    int threads = Nt * Kc * 32;
    gcn_pack_w<<<cdiv(threads, 256), 256, 0, s>>>(W, out, K, N, Nt, Kc);
}

extern "C" void kernel_launch(void* const* d_in, const int* in_sizes, int n_in,
                              void* d_out, int out_size, void* d_ws, size_t ws_size,
                              hipStream_t stream) {
    const float* x     = (const float*)d_in[0];
    const int*   eidx  = (const int*)d_in[1];
    const int*   batch = (const int*)d_in[2];
    const float* W1 = (const float*)d_in[3];  const float* b1 = (const float*)d_in[4];
    const float* W2 = (const float*)d_in[5];  const float* b2 = (const float*)d_in[6];
    const float* W3 = (const float*)d_in[7];  const float* b3 = (const float*)d_in[8];
    const float* W4 = (const float*)d_in[9];  const float* b4 = (const float*)d_in[10];
    const float* fcW1 = (const float*)d_in[11]; const float* fcb1 = (const float*)d_in[12];
    const float* fcW2 = (const float*)d_in[13]; const float* fcb2 = (const float*)d_in[14];

    const int Nn = N_NODES, E = N_EDGES;
    const int* src = eidx;
    const int* dst = eidx + E;

    // ---- workspace layout (floats) ----
    const size_t NB = (size_t)Nn * 256;
    float* ws   = (float*)d_ws;
    float* dinv = ws;                       // Nn
    float* buf0 = ws + Nn;                  // NB
    float* buf1 = buf0 + NB;                // NB
    float* buf2 = buf1 + NB;                // NB
    float* sums = buf2 + NB;                // 256*256
    float* cnt  = sums + N_GRAPHS * 256;    // 256
    float* gbuf = cnt + N_GRAPHS;           // 256*256
    float* fc1o = gbuf + N_GRAPHS * 256;    // 256*128
    // packed f16 weights (32B-aligned region)
    _Float16* pW2  = (_Float16*)(fc1o + N_GRAPHS * 128);
    _Float16* pW3  = pW2  + 2  * 1 * 32 * 16;   //  32x32
    _Float16* pW4  = pW3  + 8  * 1 * 32 * 16;   //  32x128
    _Float16* pFC1 = pW4  + 16 * 4 * 32 * 16;   // 128x256
    _Float16* pFC2 = pFC1 + 8  * 8 * 32 * 16;   // 256x128 ; pFC2: 128x10 -> 1*4*32*16
    float* outf = (float*)d_out;                // [256,10]

    // ---- pack weights into WMMA B layout ----
    launch_pack(W2,   pW2,  32,  32,  stream);
    launch_pack(W3,   pW3,  32,  128, stream);
    launch_pack(W4,   pW4,  128, 256, stream);
    launch_pack(fcW1, pFC1, 256, 128, stream);
    launch_pack(fcW2, pFC2, 128, 10,  stream);

    // ---- degree (with self loop) -> dinv ----
    gcn_fill<<<cdiv(Nn, 256), 256, 0, stream>>>(dinv, 1.0f, (size_t)Nn);
    gcn_deg<<<cdiv(E, 256), 256, 0, stream>>>(dst, dinv, E);
    gcn_rsqrt<<<cdiv(Nn, 256), 256, 0, stream>>>(dinv, Nn);

    // ---- layer 1: x[Nn,1] -> 32 ----   h=buf0, agg=buf1 -> x2=buf1
    gcn_l1_proj<<<cdiv((long long)Nn * 32, 256), 256, 0, stream>>>(x, W1, buf0, Nn);
    gcn_fill<<<cdiv((long long)Nn * 32, 256), 256, 0, stream>>>(buf1, 0.f, (size_t)Nn * 32);
    gcn_scatter<<<cdiv((long long)E << 3, 256), 256, 0, stream>>>(buf0, src, dst, dinv, buf1, E, 3, 32);
    gcn_combine<<<cdiv((long long)Nn * 32, 256), 256, 0, stream>>>(buf1, buf0, dinv, b1, Nn, 5);

    // ---- layer 2: 32 -> 32 ----        x=buf1, h=buf2, agg=buf0 -> x3=buf0
    launch_gemm(buf1, pW2, nullptr, buf2, Nn, 32, 32, 0, stream);
    gcn_fill<<<cdiv((long long)Nn * 32, 256), 256, 0, stream>>>(buf0, 0.f, (size_t)Nn * 32);
    gcn_scatter<<<cdiv((long long)E << 3, 256), 256, 0, stream>>>(buf2, src, dst, dinv, buf0, E, 3, 32);
    gcn_combine<<<cdiv((long long)Nn * 32, 256), 256, 0, stream>>>(buf0, buf2, dinv, b2, Nn, 5);

    // ---- layer 3: 32 -> 128 ----       x=buf0, h=buf1, agg=buf2 -> x4=buf2
    launch_gemm(buf0, pW3, nullptr, buf1, Nn, 32, 128, 0, stream);
    gcn_fill<<<cdiv((long long)Nn * 128, 256), 256, 0, stream>>>(buf2, 0.f, (size_t)Nn * 128);
    gcn_scatter<<<cdiv((long long)E << 5, 256), 256, 0, stream>>>(buf1, src, dst, dinv, buf2, E, 5, 128);
    gcn_combine<<<cdiv((long long)Nn * 128, 256), 256, 0, stream>>>(buf2, buf1, dinv, b3, Nn, 7);

    // ---- layer 4: 128 -> 256 ----      x=buf2, h=buf0, agg=buf1 -> last=buf1
    launch_gemm(buf2, pW4, nullptr, buf0, Nn, 128, 256, 0, stream);
    gcn_fill<<<cdiv((long long)Nn * 256, 256), 256, 0, stream>>>(buf1, 0.f, (size_t)Nn * 256);
    gcn_scatter<<<cdiv((long long)E << 6, 256), 256, 0, stream>>>(buf0, src, dst, dinv, buf1, E, 6, 256);
    gcn_combine<<<cdiv((long long)Nn * 256, 256), 256, 0, stream>>>(buf1, buf0, dinv, b4, Nn, 8);

    // ---- global mean pool ----
    gcn_fill<<<cdiv(N_GRAPHS * 256 + N_GRAPHS, 256), 256, 0, stream>>>(
        sums, 0.f, (size_t)(N_GRAPHS * 256 + N_GRAPHS));
    gcn_pool_accum<<<cdiv((long long)Nn * 64, 256), 256, 0, stream>>>(buf1, batch, sums, Nn);
    gcn_pool_cnt<<<cdiv(Nn, 256), 256, 0, stream>>>(batch, cnt, Nn);
    gcn_pool_mean<<<cdiv(N_GRAPHS * 256, 256), 256, 0, stream>>>(sums, cnt, gbuf);

    // ---- FC head (WMMA) ----
    launch_gemm(gbuf, pFC1, fcb1, fc1o, N_GRAPHS, 256, 128, 1, stream);
    launch_gemm(fc1o, pFC2, fcb2, outf, N_GRAPHS, 128, 10, 0, stream);
}
